// WindowAttentionGeo_35098472743142
// MI455X (gfx1250) — matrix operand order repaired and checked by reading, not compiled, not run on hardware
//
#include <hip/hip_runtime.h>
#include <stdint.h>

typedef __attribute__((ext_vector_type(16))) _Float16 v16h;
typedef __attribute__((ext_vector_type(8)))  float    v8f;
typedef __attribute__((ext_vector_type(4)))  int      v4i;

#define DIMC 180
#define NH   6
#define HD   30
#define HDP  32
#define NTOK 256
#define NWIN 64
#define XK   192          // padded K for projection (6 chunks of 32)
#define WST  196          // f16 W row stride: bank step 98%64=34 -> conflict-free
#define WH_BYTES (192 * WST * 2)

#if defined(__has_builtin)
#if __has_builtin(__builtin_amdgcn_global_load_async_to_lds_b128) && \
    __has_builtin(__builtin_amdgcn_s_wait_asynccnt)
#define HAVE_ASYNC_LDS 1
#endif
#endif

union FragA { uint32_t u[8]; v16h v; };

static __device__ __forceinline__ v8f wmma16(const FragA& a, const FragA& b, v8f c) {
    return __builtin_amdgcn_wmma_f32_16x16x32_f16(false, a.v, false, b.v,
                                                  (short)0, c, false, false);
}

// ---------------------------------------------------------------------------
// Kernel 0: prep.  (a) densify relative-position bias: rpbD[h][q*256+k]
//                  (b) convert proj W to f16, padded [192][WST]
// ---------------------------------------------------------------------------
__global__ __launch_bounds__(256)
void prep_kernel(const int* __restrict__ rpi, const float* __restrict__ rpb,
                 const float* __restrict__ W,
                 float* __restrict__ rpbD, _Float16* __restrict__ Wh)
{
    const int idx = blockIdx.x * 256 + threadIdx.x;
    const int total1 = NH * NTOK * NTOK;               // 393216
    if (idx < total1) {
        int h = idx >> 16;                             // / 65536
        int e = idx & 65535;
        rpbD[idx] = rpb[rpi[e] * NH + h];
    } else {
        int j = idx - total1;
        if (j < 192 * WST) {
            int n = j / WST, k = j % WST;
            float w = (n < DIMC && k < DIMC) ? W[n * DIMC + k] : 0.f;
            Wh[j] = (_Float16)w;
        }
    }
}

// ---------------------------------------------------------------------------
// Kernel 1: per (head, window) attention.  block = 256 thr (8 waves, wave32)
// ---------------------------------------------------------------------------
__global__ __launch_bounds__(256)
void attn_kernel(const float* __restrict__ qkv,
                 const float* __restrict__ maskp, const float* __restrict__ geo,
                 const float* __restrict__ rpbD, const float* __restrict__ gscale,
                 _Float16* __restrict__ X)
{
    const int h    = blockIdx.x;
    const int b    = blockIdx.y;
    const int tid  = threadIdx.x;
    const int lane = tid & 31;
    const int wave = tid >> 5;
    const int ml   = lane & 15;   // column / row-in-tile lane id
    const int hi   = lane >> 4;   // which half of the wave

    __shared__ _Float16 Qs[NTOK][HDP];        // 16 KB, row-major [tok][d]
    __shared__ _Float16 Ks[NTOK][HDP];        // 16 KB, row-major [tok][d]
    __shared__ _Float16 Vt[HDP][NTOK + 8];    // 16.9 KB, transposed [d][tok]
    __shared__ _Float16 Ps[8][16][36];        // 9.2 KB per-wave P staging

    const float qscale = 0.18257418583505536f;   // 30^-0.5
    for (int idx = tid; idx < NTOK * HDP; idx += 256) {
        int n = idx >> 5, d = idx & 31;
        const float* base = qkv + (size_t)(b * NTOK + n) * (3 * DIMC) + h * HD;
        float qv = 0.f, kv = 0.f, vv = 0.f;
        if (d < HD) { qv = base[d]; kv = base[DIMC + d]; vv = base[2 * DIMC + d]; }
        Qs[n][d] = (_Float16)(qv * qscale);
        Ks[n][d] = (_Float16)kv;
        Vt[d][n] = (_Float16)vv;
    }
    __syncthreads();

    const float  gs    = gscale[h];
    const float* geob  = geo  + (size_t)b * NTOK * NTOK;
    const float* maskb = maskp + (size_t)(b & (NWIN - 1)) * NTOK * NTOK;
    const float* rpbh  = rpbD + (size_t)h * NTOK * NTOK;
    const v8f    zero  = {0.f, 0.f, 0.f, 0.f, 0.f, 0.f, 0.f, 0.f};

    for (int rbi = 0; rbi < 2; ++rbi) {
        const int row0 = (wave + rbi * 8) * 16;

        // ---- A fragment: 16x32 rows of Q ----
        FragA qa;
        #pragma unroll
        for (int j = 0; j < 8; ++j) {
            int kk = hi * 16 + 2 * j;
            qa.u[j] = *(const uint32_t*)&Qs[row0 + ml][kk];
        }

        // ---- S = Q * K^T : one 16x256 row of scores, 16 WMMA tiles ----
        v8f acc[16];
        #pragma unroll
        for (int ct = 0; ct < 16; ++ct) {
            FragA kb;
            #pragma unroll
            for (int j = 0; j < 8; ++j)
                kb.u[j] = *(const uint32_t*)&Ks[ct * 16 + ml][hi * 16 + 2 * j];
            acc[ct] = wmma16(qa, kb, zero);
        }

        // ---- fused bias: dense rpbD - geo*scale + shift-window mask ----
        #pragma unroll
        for (int ct = 0; ct < 16; ++ct) {
            int ncol = ct * 16 + ml;
            #pragma unroll
            for (int r = 0; r < 8; ++r) {
                int mrow = row0 + r + hi * 8;
                size_t e = (size_t)mrow * NTOK + ncol;
                float bias = rpbh[e] - geob[e] * gs + maskb[e];
                acc[ct][r] += bias;
            }
        }

        // ---- softmax: lanes 0-15 hold row r, lanes 16-31 hold row r+8 ----
        #pragma unroll
        for (int r = 0; r < 8; ++r) {
            float mx = -3.402823466e38f;
            #pragma unroll
            for (int ct = 0; ct < 16; ++ct) mx = fmaxf(mx, acc[ct][r]);
            mx = fmaxf(mx, __shfl_xor(mx, 1));
            mx = fmaxf(mx, __shfl_xor(mx, 2));
            mx = fmaxf(mx, __shfl_xor(mx, 4));
            mx = fmaxf(mx, __shfl_xor(mx, 8));
            float s = 0.f;
            #pragma unroll
            for (int ct = 0; ct < 16; ++ct) {
                float ev = __expf(acc[ct][r] - mx);
                acc[ct][r] = ev;
                s += ev;
            }
            s += __shfl_xor(s, 1);
            s += __shfl_xor(s, 2);
            s += __shfl_xor(s, 4);
            s += __shfl_xor(s, 8);
            float inv = __builtin_amdgcn_rcpf(s);
            #pragma unroll
            for (int ct = 0; ct < 16; ++ct) acc[ct][r] *= inv;
        }

        // ---- O = P * V : K loop over 8 chunks of 32 keys ----
        v8f oacc0 = zero, oacc1 = zero;
        for (int kc = 0; kc < 8; ++kc) {
            // restage two P tiles (C layout -> row-major) through per-wave LDS
            #pragma unroll
            for (int t2 = 0; t2 < 2; ++t2) {
                int ct = kc * 2 + t2;
                #pragma unroll
                for (int r = 0; r < 8; ++r)
                    Ps[wave][r + hi * 8][t2 * 16 + ml] = (_Float16)acc[ct][r];
            }
            // A fragment of P (wave-private region; DS ops are in-order)
            FragA pa;
            #pragma unroll
            for (int j = 0; j < 8; ++j) {
                int kk = hi * 16 + 2 * j;
                pa.u[j] = *(const uint32_t*)&Ps[wave][ml][kk];
            }
            // B fragments of V (two 16-wide dim tiles), key pairs contiguous in Vt
            FragA vb0, vb1;
            #pragma unroll
            for (int j = 0; j < 8; ++j) {
                int key = kc * 32 + hi * 16 + 2 * j;
                vb0.u[j] = *(const uint32_t*)&Vt[ml][key];
                vb1.u[j] = *(const uint32_t*)&Vt[16 + ml][key];
            }
            oacc0 = wmma16(pa, vb0, oacc0);
            oacc1 = wmma16(pa, vb1, oacc1);
        }

        // ---- write per-head slice of X as f16 ----
        #pragma unroll
        for (int t2 = 0; t2 < 2; ++t2) {
            v8f oa = t2 ? oacc1 : oacc0;
            int d = t2 * 16 + ml;
            if (d < HD) {
                #pragma unroll
                for (int r = 0; r < 8; ++r) {
                    int mrow = row0 + r + hi * 8;
                    X[(size_t)(b * NTOK + mrow) * XK + h * HD + d] = (_Float16)oa[r];
                }
            }
        }
    }

    // one head zero-fills the K padding [180..192) of this window's rows
    if (h == NH - 1) {
        for (int idx = tid; idx < NTOK * (XK - DIMC); idx += 256) {
            int n = idx / (XK - DIMC);
            int d = DIMC + idx % (XK - DIMC);
            X[(size_t)(b * NTOK + n) * XK + d] = (_Float16)0.f;
        }
    }
}

// ---------------------------------------------------------------------------
// Kernel 2: out = X @ W^T + bias.  M=131072, N=180(->192), K=180(->192)
// block = 256 thr (8 waves); wave handles 16 rows x 12 N-tiles.
// W (f16, padded) staged in LDS via CDNA5 async global->LDS copy.
// ---------------------------------------------------------------------------
__global__ __launch_bounds__(256)
void proj_kernel(const _Float16* __restrict__ X, const _Float16* __restrict__ Wh,
                 const float* __restrict__ bias, float* __restrict__ out)
{
    extern __shared__ __align__(16) char smem[];
    _Float16* Wlds = (_Float16*)smem;

    const int tid  = threadIdx.x;
    const int lane = tid & 31;
    const int wave = tid >> 5;
    const int ml   = lane & 15;
    const int hi   = lane >> 4;
    const size_t row0 = (size_t)blockIdx.x * 128 + (size_t)wave * 16;

    // ---- stage padded f16 W [192][WST] into LDS ----
#if defined(HAVE_ASYNC_LDS)
    for (int off = tid * 16; off < WH_BYTES; off += 256 * 16) {
        __attribute__((address_space(1))) v4i* gp =
            (__attribute__((address_space(1))) v4i*)((char*)Wh + off);
        __attribute__((address_space(3))) v4i* lp =
            (__attribute__((address_space(3))) v4i*)(smem + off);
        __builtin_amdgcn_global_load_async_to_lds_b128(gp, lp, 0, 0);
    }
    __builtin_amdgcn_s_wait_asynccnt(0);
#else
    for (int off = tid; off < WH_BYTES / 16; off += 256)
        ((uint4*)smem)[off] = ((const uint4*)Wh)[off];
#endif
    __syncthreads();

    const v8f zero = {0.f, 0.f, 0.f, 0.f, 0.f, 0.f, 0.f, 0.f};
    v8f acc[12];
    #pragma unroll
    for (int nt = 0; nt < 12; ++nt) acc[nt] = zero;

    for (int kc = 0; kc < 6; ++kc) {
        FragA a;
        #pragma unroll
        for (int j = 0; j < 8; ++j) {
            int kk = kc * 32 + hi * 16 + 2 * j;
            a.u[j] = *(const uint32_t*)&X[(row0 + ml) * XK + kk];
        }
        #pragma unroll
        for (int nt = 0; nt < 12; ++nt) {
            int n = nt * 16 + ml;
            FragA bf;
            #pragma unroll
            for (int j = 0; j < 8; ++j) {
                int kk = kc * 32 + hi * 16 + 2 * j;
                bf.u[j] = *(const uint32_t*)&Wlds[n * WST + kk];
            }
            acc[nt] = wmma16(a, bf, acc[nt]);
        }
    }

    #pragma unroll
    for (int nt = 0; nt < 12; ++nt) {
        int n = nt * 16 + ml;
        if (n < DIMC) {
            float bb = bias[n];
            #pragma unroll
            for (int r = 0; r < 8; ++r) {
                size_t row = row0 + (size_t)(r + hi * 8);
                out[row * DIMC + n] = acc[nt][r] + bb;
            }
        }
    }
}

// ---------------------------------------------------------------------------
extern "C" void kernel_launch(void* const* d_in, const int* in_sizes, int n_in,
                              void* d_out, int out_size, void* d_ws, size_t ws_size,
                              hipStream_t stream) {
    (void)n_in; (void)out_size; (void)ws_size;
    const float* qkv    = (const float*)d_in[0];
    const int*   rpi    = (const int*)  d_in[1];
    const float* maskp  = (const float*)d_in[2];
    const float* geo    = (const float*)d_in[3];
    const float* rpb    = (const float*)d_in[4];
    const float* gscale = (const float*)d_in[5];
    const float* projw  = (const float*)d_in[6];
    const float* projb  = (const float*)d_in[7];
    float* out = (float*)d_out;

    const int B    = in_sizes[0] / (NTOK * 3 * DIMC);   // 512
    const int rows = B * NTOK;                           // 131072

    // workspace layout
    char* ws = (char*)d_ws;
    _Float16* X    = (_Float16*)ws;                                   // rows*192 f16
    float*    rpbD = (float*)(ws + (size_t)rows * XK * sizeof(_Float16));
    _Float16* Wh   = (_Float16*)((char*)rpbD + (size_t)NH * NTOK * NTOK * sizeof(float));

    const int prep_elems = NH * NTOK * NTOK + 192 * WST;
    prep_kernel<<<(prep_elems + 255) / 256, 256, 0, stream>>>(rpi, rpb, projw, rpbD, Wh);

    dim3 g1(NH, B);
    attn_kernel<<<g1, 256, 0, stream>>>(qkv, maskp, geo, rpbD, gscale, X);

    proj_kernel<<<rows / 128, 256, WH_BYTES, stream>>>(X, Wh, projb, out);
}
